// SimplifiedMamba2Block_670014898335
// MI455X (gfx1250) — matrix operand order repaired and checked by reading, not compiled, not run on hardware
//
#include <hip/hip_runtime.h>
#include <hip/hip_bf16.h>

// ---------------------------------------------------------------------------
// Simplified Mamba2 block on MI455X (gfx1250, wave32, WMMA + async-to-LDS).
// Pipeline: cvt(bf16) -> WMMA GEMM1(split) -> conv+silu -> xproj(softplus/B)
//           -> selective scan -> gate+cvt -> WMMA GEMM3 -> out
// ---------------------------------------------------------------------------

typedef __attribute__((ext_vector_type(16))) __bf16 v16bf;
typedef __attribute__((ext_vector_type(8)))  __bf16 v8bf;
typedef __attribute__((ext_vector_type(8)))  float  v8f;

constexpr int B_  = 2;
constexpr int S_  = 4096;
constexpr int DM_ = 768;
constexpr int DS_ = 16;
constexpr int DC_ = 4;
constexpr int DI_ = DM_ * 2;     // 1536
constexpr int M_  = B_ * S_;     // 8192 rows
constexpr int N1_ = 2 * DI_;     // 3072
constexpr int K1_ = DM_;         // 768
constexpr int N3_ = DM_;         // 768
constexpr int K3_ = DI_;         // 1536

// ---------------------------------------------------------------------------
// CDNA5 async global->LDS copy (GLOBAL_LOAD_ASYNC_TO_LDS_B128, ASYNCcnt).
// LDS dest VGPR holds the LDS byte offset = low 32 bits of the generic pointer
// (ISA 10.2: the LDS aperture uses the address's low 32 bits).
// ---------------------------------------------------------------------------
__device__ __forceinline__ void async_copy_b128(void* lds_dst, const void* gsrc) {
    unsigned lds_off = (unsigned)(unsigned long long)lds_dst;
    asm volatile("global_load_async_to_lds_b128 %0, %1, off"
                 :: "v"(lds_off), "v"(gsrc) : "memory");
}

#if defined(__has_builtin)
#if __has_builtin(__builtin_amdgcn_s_wait_asynccnt)
#define WAIT_ASYNC(n) __builtin_amdgcn_s_wait_asynccnt(n)
#endif
#endif
#ifndef WAIT_ASYNC
#define WAIT_ASYNC(n) asm volatile("s_wait_asynccnt %0" :: "n"(n) : "memory")
#endif

// ---------------------------------------------------------------------------
// f32 -> bf16 conversion
// ---------------------------------------------------------------------------
__global__ void cvt_bf16_kernel(const float* __restrict__ in, __bf16* __restrict__ out, int n) {
    int i = blockIdx.x * blockDim.x + threadIdx.x;
    if (i < n) out[i] = (__bf16)in[i];
}

// ---------------------------------------------------------------------------
// WMMA bf16 GEMM:  C[m,n] = sum_k A[m,k] * W[n,k] + bias[n]
//   A: (M,K) bf16 row-major, W: (N,K) bf16 row-major (i.e. computes A @ W^T)
// Block tile 128x128, BK=32; double-buffered LDS fed by async-to-LDS copies.
// 8 waves: 4(M) x 2(N), each wave 32x64 = 2x4 v_wmma_f32_16x16x32_bf16 tiles.
// SPLIT=true splits columns at splitN into outA / outB (GEMM1 path).
// ---------------------------------------------------------------------------
template <bool SPLIT>
__global__ __launch_bounds__(256)
void wmma_gemm_bf16(const __bf16* __restrict__ A, const __bf16* __restrict__ W,
                    const float* __restrict__ bias,
                    float* __restrict__ outA, float* __restrict__ outB,
                    int M, int N, int K, int splitN) {
    constexpr int BM = 128, BN = 128, BK = 32;
    constexpr int LDT = 40;                     // LDS row stride (elems): 80B, conflict-padded
    __shared__ __align__(16) __bf16 At[2][BM * LDT];
    __shared__ __align__(16) __bf16 Bt[2][BN * LDT];

    const int tid  = threadIdx.x;
    const int lane = tid & 31;
    const int wave = tid >> 5;
    const int wm   = (wave & 3) * 32;           // wave M offset in block tile
    const int wn   = (wave >> 2) * 64;          // wave N offset in block tile
    const int m0   = blockIdx.y * BM;
    const int n0   = blockIdx.x * BN;
    const int h    = lane >> 4;                 // lane half (ISA 16-bit layouts)
    const int ln   = lane & 15;

    // Straight-line staging addressing: 512 x 16B chunks per tile, 2 per thread.
    const int r0 = tid >> 2,        c0 = tid & 3;          // chunk tid
    const int r1 = (tid + 256) >> 2, c1 = (tid + 256) & 3; // chunk tid+256

    v8f acc[2][4];
#pragma unroll
    for (int i = 0; i < 2; ++i)
#pragma unroll
        for (int j = 0; j < 4; ++j)
#pragma unroll
            for (int v = 0; v < 8; ++v) acc[i][j][v] = 0.f;

    // Issue one tile's 4 async copies (per thread / per wave): A x2, B x2.
    auto issue_tile = [&](int buf, int k0) {
        async_copy_b128(&At[buf][r0 * LDT + c0 * 8], A + (size_t)(m0 + r0) * K + k0 + c0 * 8);
        async_copy_b128(&At[buf][r1 * LDT + c1 * 8], A + (size_t)(m0 + r1) * K + k0 + c1 * 8);
        async_copy_b128(&Bt[buf][r0 * LDT + c0 * 8], W + (size_t)(n0 + r0) * K + k0 + c0 * 8);
        async_copy_b128(&Bt[buf][r1 * LDT + c1 * 8], W + (size_t)(n0 + r1) * K + k0 + c1 * 8);
    };

    const int nk = K / BK;
    issue_tile(0, 0);

    for (int kk = 0; kk < nk; ++kk) {
        const int cur = kk & 1;
        if (kk + 1 < nk) {
            issue_tile(cur ^ 1, (kk + 1) * BK);
            WAIT_ASYNC(4);     // in-order: the 4 newest are next tile; current tile done
        } else {
            WAIT_ASYNC(0);
        }
        __syncthreads();       // current tile visible to all waves

        // A fragments: 16x32 bf16, lane ln = row; h=0 -> K[0..7],K[16..23],
        // h=1 -> K[8..15],K[24..31]  (ISA 16-bit A layout).
        v16bf af[2];
#pragma unroll
        for (int i = 0; i < 2; ++i) {
            const __bf16* base = &At[cur][(wm + i * 16 + ln) * LDT];
            v8bf lo = *(const v8bf*)(base + 8 * h);
            v8bf hi = *(const v8bf*)(base + 16 + 8 * h);
#pragma unroll
            for (int e = 0; e < 8; ++e) { af[i][e] = lo[e]; af[i][8 + e] = hi[e]; }
        }
        // B fragments: 32x16 bf16, lane ln = column n; h=0 -> K[0..7],K[16..23], h=1 -> K[8..15],K[24..31].
        v16bf bfr[4];
#pragma unroll
        for (int j = 0; j < 4; ++j) {
            const __bf16* base = &Bt[cur][(wn + j * 16 + ln) * LDT];
            v8bf lo = *(const v8bf*)(base + 8 * h);
            v8bf hi = *(const v8bf*)(base + 16 + 8 * h);
#pragma unroll
            for (int e = 0; e < 8; ++e) { bfr[j][e] = lo[e]; bfr[j][8 + e] = hi[e]; }
        }

#pragma unroll
        for (int i = 0; i < 2; ++i)
#pragma unroll
            for (int j = 0; j < 4; ++j)
                acc[i][j] = __builtin_amdgcn_wmma_f32_16x16x32_bf16(
                    false, af[i], false, bfr[j], (short)0, acc[i][j], false, false);
        __syncthreads();       // done reading 'cur' before it is refilled next iter
    }

    // Epilogue. f32 C/D layout: VGPR v -> row v + 8*h, lane ln -> column.
#pragma unroll
    for (int i = 0; i < 2; ++i)
#pragma unroll
        for (int j = 0; j < 4; ++j)
#pragma unroll
            for (int v = 0; v < 8; ++v) {
                int row = m0 + wm + i * 16 + v + 8 * h;
                int col = n0 + wn + j * 16 + ln;
                float val = acc[i][j][v] + bias[col];
                if (SPLIT) {
                    if (col < splitN) outA[(size_t)row * splitN + col] = val;
                    else              outB[(size_t)row * splitN + (col - splitN)] = val;
                } else {
                    outA[(size_t)row * N + col] = val;
                }
            }
}

// ---------------------------------------------------------------------------
// Depthwise causal conv1d (k=4) + SiLU.  xi: (B,S,DI) f32 -> xc: (B,S,DI)
// ---------------------------------------------------------------------------
__global__ void conv_silu_kernel(const float* __restrict__ xi, const float* __restrict__ cw,
                                 const float* __restrict__ cb, float* __restrict__ xc) {
    int idx = blockIdx.x * blockDim.x + threadIdx.x;      // over B*S*DI (fits int)
    int d = idx % DI_;
    int bt = idx / DI_;
    int t = bt % S_;
    int b = bt / S_;
    float acc = cb[d];
#pragma unroll
    for (int j = 0; j < DC_; ++j) {
        int tt = t - (DC_ - 1) + j;
        if (tt >= 0)
            acc = fmaf(xi[((size_t)b * S_ + tt) * DI_ + d], cw[d * DC_ + j], acc);
    }
    xc[idx] = acc / (1.f + __expf(-acc));                 // silu
}

// ---------------------------------------------------------------------------
// x-projection: ssm[row, n] = sum_k xc[row,k] * W_xproj[n,k] + b[n], n in [0,32)
// One wave per row; lane owns one output column; x broadcast via __shfl (wave32).
// Fused epilogue: delta = softplus(ssm[:,0:16]); Bm = ssm[:,16:32].
// ---------------------------------------------------------------------------
__global__ __launch_bounds__(256)
void xproj_kernel(const float* __restrict__ xc, const float* __restrict__ W,
                  const float* __restrict__ bias,
                  float* __restrict__ delta, float* __restrict__ Bm) {
    int row  = blockIdx.x * 8 + (threadIdx.x >> 5);
    int lane = threadIdx.x & 31;
    const float* xrow = xc + (size_t)row * K3_;
    const float* wrow = W + (size_t)lane * K3_;
    float acc = bias[lane];
    for (int k0 = 0; k0 < K3_; k0 += 32) {
        float xv = xrow[k0 + lane];
#pragma unroll
        for (int j = 0; j < 32; ++j) {
            float xs = __shfl(xv, j, 32);
            acc = fmaf(xs, wrow[k0 + j], acc);
        }
    }
    if (lane < DS_) {
        float sp = (acc > 30.f) ? acc : log1pf(__expf(acc));   // softplus
        delta[(size_t)row * DS_ + lane] = sp;
    } else {
        Bm[(size_t)row * DS_ + (lane - DS_)] = acc;
    }
}

// ---------------------------------------------------------------------------
// Selective scan. One thread per (b, d) channel; 16-wide state in registers.
// delta/B staged per 128-step chunk in LDS (broadcast reads, conflict-free).
// ---------------------------------------------------------------------------
__global__ __launch_bounds__(128)
void scan_kernel(const float* __restrict__ xc, const float* __restrict__ delta,
                 const float* __restrict__ Bm, const float* __restrict__ A_log,
                 const float* __restrict__ Dvec, float* __restrict__ y) {
    constexpr int TCH = 128;
    __shared__ float dl[TCH * DS_];
    __shared__ float bl[TCH * DS_];
    const int d = blockIdx.x * 128 + threadIdx.x;
    const int b = blockIdx.y;

    float a[DS_], h[DS_];
#pragma unroll
    for (int s = 0; s < DS_; ++s) {
        a[s] = -__expf(A_log[d * DS_ + s]);
        h[s] = 0.f;
    }
    const float Dd = Dvec[d];

    for (int t0 = 0; t0 < S_; t0 += TCH) {
        for (int i = threadIdx.x; i < TCH * DS_; i += 128) {
            dl[i] = delta[((size_t)b * S_ + t0) * DS_ + i];
            bl[i] = Bm[((size_t)b * S_ + t0) * DS_ + i];
        }
        __syncthreads();
        for (int t = 0; t < TCH; ++t) {
            const float x = xc[((size_t)b * S_ + t0 + t) * DI_ + d];
            float yv = Dd * x;
#pragma unroll
            for (int s = 0; s < DS_; ++s) {
                float Bv = bl[t * DS_ + s];
                h[s] = h[s] * __expf(a[s] * dl[t * DS_ + s]) + x * Bv;
                yv = fmaf(h[s], Bv, yv);
            }
            y[((size_t)b * S_ + t0 + t) * DI_ + d] = yv;
        }
        __syncthreads();
    }
}

// ---------------------------------------------------------------------------
// Gate: out_bf16 = (y * silu(res)) as bf16 (A-operand of GEMM3).
// ---------------------------------------------------------------------------
__global__ void gate_cvt_kernel(const float* __restrict__ y, const float* __restrict__ res,
                                __bf16* __restrict__ out, int n) {
    int i = blockIdx.x * blockDim.x + threadIdx.x;
    if (i < n) {
        float r = res[i];
        float g = r / (1.f + __expf(-r));
        out[i] = (__bf16)(y[i] * g);
    }
}

// ---------------------------------------------------------------------------
extern "C" void kernel_launch(void* const* d_in, const int* in_sizes, int n_in,
                              void* d_out, int out_size, void* d_ws, size_t ws_size,
                              hipStream_t stream) {
    const float* x      = (const float*)d_in[0];
    const float* W_in   = (const float*)d_in[1];
    const float* b_in   = (const float*)d_in[2];
    const float* conv_w = (const float*)d_in[3];
    const float* conv_b = (const float*)d_in[4];
    const float* W_xp   = (const float*)d_in[5];
    const float* b_xp   = (const float*)d_in[6];
    const float* A_log  = (const float*)d_in[7];
    const float* Dv     = (const float*)d_in[8];
    const float* W_out  = (const float*)d_in[9];
    const float* b_out  = (const float*)d_in[10];
    float* out = (float*)d_out;

    char* ws = (char*)d_ws;
    size_t off = 0;
    auto alloc = [&](size_t bytes) -> void* {
        void* p = ws + off;
        off = (off + bytes + 255) & ~(size_t)255;
        return p;
    };
    __bf16* xb    = (__bf16*)alloc((size_t)M_ * K1_ * sizeof(__bf16));
    __bf16* Wib   = (__bf16*)alloc((size_t)N1_ * K1_ * sizeof(__bf16));
    __bf16* Wob   = (__bf16*)alloc((size_t)N3_ * K3_ * sizeof(__bf16));
    float*  xi    = (float*)alloc((size_t)M_ * DI_ * sizeof(float));   // reused as yscan
    float*  res   = (float*)alloc((size_t)M_ * DI_ * sizeof(float));
    float*  xc    = (float*)alloc((size_t)M_ * DI_ * sizeof(float));
    float*  delta = (float*)alloc((size_t)M_ * DS_ * sizeof(float));
    float*  Bm    = (float*)alloc((size_t)M_ * DS_ * sizeof(float));
    __bf16* yb    = (__bf16*)alloc((size_t)M_ * DI_ * sizeof(__bf16));
    float*  yscan = xi;   // xi is dead after the conv kernel

    // 1) bf16 conversions
    {
        int n = M_ * K1_;
        cvt_bf16_kernel<<<(n + 255) / 256, 256, 0, stream>>>(x, xb, n);
        n = N1_ * K1_;
        cvt_bf16_kernel<<<(n + 255) / 256, 256, 0, stream>>>(W_in, Wib, n);
        n = N3_ * K3_;
        cvt_bf16_kernel<<<(n + 255) / 256, 256, 0, stream>>>(W_out, Wob, n);
    }
    // 2) GEMM1: xr = x @ W_in^T + b_in, split into xi / res
    {
        dim3 grid(N1_ / 128, M_ / 128);
        wmma_gemm_bf16<true><<<grid, 256, 0, stream>>>(xb, Wib, b_in, xi, res,
                                                       M_, N1_, K1_, DI_);
    }
    // 3) depthwise conv + silu
    {
        int n = M_ * DI_;
        conv_silu_kernel<<<(n + 255) / 256, 256, 0, stream>>>(xi, conv_w, conv_b, xc);
    }
    // 4) x-projection + softplus split
    {
        xproj_kernel<<<M_ / 8, 256, 0, stream>>>(xc, W_xp, b_xp, delta, Bm);
    }
    // 5) selective scan
    {
        dim3 grid(DI_ / 128, B_);
        scan_kernel<<<grid, 128, 0, stream>>>(xc, delta, Bm, A_log, Dv, yscan);
    }
    // 6) gate + bf16 convert
    {
        int n = M_ * DI_;
        gate_cvt_kernel<<<(n + 255) / 256, 256, 0, stream>>>(yscan, res, yb, n);
    }
    // 7) GEMM3: out = (y*silu(res)) @ W_out^T + b_out
    {
        dim3 grid(N3_ / 128, M_ / 128);
        wmma_gemm_bf16<false><<<grid, 256, 0, stream>>>(yb, Wob, b_out, out, nullptr,
                                                        M_, N3_, K3_, 0);
    }
}
